// SuperONN1d_30185030157072
// MI455X (gfx1250) — matrix operand order repaired and checked by reading, not compiled, not run on hardware
//
#include <hip/hip_runtime.h>

// ---------------------------------------------------------------------------
// SuperONN1d forward on gfx1250 (MI455X):
//   shifted-lerp channel gather -> powers 1..3 (bf16) -> im2col GEMM via
//   V_WMMA_F32_16X16X32_BF16 with f32 accumulation.
// Problem: B=16, Cin=128, L=8192, Cout=128, K=9, Q=3, PAD=4.
//
// Wave tiling: 8 waves = 4(M) x 2(N); each wave owns 32x32 of the 128x64
// output tile as 2x2 WMMA accumulators -> 2-way register reuse of both A and
// B fragments, halving LDS B-feed bandwidth per WMMA vs. an M-only split.
// ---------------------------------------------------------------------------

typedef __attribute__((ext_vector_type(16))) __bf16       v16bf;
typedef __attribute__((ext_vector_type(8)))  float        v8f;
typedef __attribute__((ext_vector_type(4)))  unsigned int v4u;

#define BATCH 16
#define CIN   128
#define LEN   8192
#define COUT  128
#define KW    9
#define QPOW  3
#define PADC  4
#define RTOT  (QPOW * CIN)        // 384 reduction rows per tap
#define NTILE 64                  // output positions per workgroup
#define PROWS (NTILE + KW - 1)    // 72 staged positions (includes halo)
#define RSTR  (RTOT + 8)          // 392 ushorts = 784B row stride (bank spread)

__device__ __forceinline__ unsigned short f2bf(float f) {
  unsigned int u = __float_as_uint(f);
  u += 0x7FFFu + ((u >> 16) & 1u);        // round-to-nearest-even
  return (unsigned short)(u >> 16);
}

// Repack weights (Cout, Q*Cin, K) f32 -> w2[k][co][r] bf16, r = q*CIN+ci.
__global__ __launch_bounds__(256) void prep_w_bf16(
    const float* __restrict__ w, unsigned short* __restrict__ w2) {
  int idx = blockIdx.x * 256 + threadIdx.x;
  if (idx >= KW * COUT * RTOT) return;
  int r  = idx % RTOT;
  int co = (idx / RTOT) % COUT;
  int k  = idx / (RTOT * COUT);
  w2[idx] = f2bf(w[((size_t)co * RTOT + r) * KW + k]);
}

__global__ __launch_bounds__(256) void superonn_wmma(
    const float* __restrict__ x, const unsigned short* __restrict__ w2,
    const float* __restrict__ bias, const float* __restrict__ shifts,
    float* __restrict__ out) {
  // Position-major bf16 staging: lds[p][r], 72 x 392 ushorts = 56,448 B.
  __shared__ unsigned short lds[PROWS * RSTR];

  const int ltile = blockIdx.x;
  const int b     = blockIdx.y;
  const int l0    = ltile * NTILE;
  const int tid   = threadIdx.x;

  // ---- Stage: fractional shift (zeros padding) + powers 1..3, to bf16 LDS.
  for (int it = tid; it < CIN * PROWS; it += 256) {
    const int ci = it / PROWS;
    const int p  = it % PROWS;
    const float s  = shifts[ci * 2] * 4.0f;           // MAX_SHIFT = 4
    const float pf = (float)(l0 + p - PADC) + s;
    const float fl = floorf(pf);
    const int   i0 = (int)fl;
    const float w1 = pf - fl;
    const float* xc = x + ((size_t)b * CIN + ci) * LEN;
    const float x0 = (i0 >= 0 && i0 < LEN) ? xc[i0] : 0.0f;
    const int   i1 = i0 + 1;
    const float x1 = (i1 >= 0 && i1 < LEN) ? xc[i1] : 0.0f;
    const float v  = x0 * (1.0f - w1) + x1 * w1;
    unsigned short* row = lds + p * RSTR;
    row[ci]           = f2bf(v);
    row[CIN + ci]     = f2bf(v * v);
    row[2 * CIN + ci] = f2bf(v * v * v);
  }
  __syncthreads();

  // ---- GEMM: 8 waves = 4(M) x 2(N); each wave owns 32(Cout) x 32(L).
  const int wave = tid >> 5;        // wave32
  const int lane = tid & 31;
  const int mrow = lane & 15;
  const int half = lane >> 4;       // 0: lanes 0-15, 1: lanes 16-31
  const int mw   = wave >> 1;       // 0..3 -> Cout strip
  const int nw   = wave & 1;        // 0..1 -> L strip
  const int co0  = mw * 32;
  const int n0w  = nw * 32;

  union Frag { v16bf v; v4u q[2]; };

  v8f acc[2][2] = {};

  for (int k = 0; k < KW; ++k) {
    const unsigned short* wk =
        w2 + (size_t)(k * COUT + co0 + mrow) * RTOT;
#pragma unroll
    for (int rc = 0; rc < RTOT / 32; ++rc) {
      // A fragments 16x32 bf16: lane<16 holds K in {h8..h8+7} u {16+h8..}.
      Frag a[2];
#pragma unroll
      for (int mi = 0; mi < 2; ++mi) {
        const unsigned short* ab =
            wk + (size_t)(mi * 16) * RTOT + rc * 32 + half * 8;
        a[mi].q[0] = *(const v4u*)(ab);       // K = rc*32 + half*8   .. +7
        a[mi].q[1] = *(const v4u*)(ab + 16);  // K = rc*32+16+half*8  .. +7
      }
      // B fragments 32x16 bf16: column N = lane%16 -> position p = n + k;
      // lane half h holds K in {8h..8h+7} u {16+8h..16+8h+7} of the chunk.
      Frag bf[2];
#pragma unroll
      for (int ni = 0; ni < 2; ++ni) {
        const int p = n0w + ni * 16 + mrow + k;
        const unsigned short* bb = lds + p * RSTR + rc * 32 + half * 8;
        bf[ni].q[0] = *(const v4u*)(bb);
        bf[ni].q[1] = *(const v4u*)(bb + 16);
      }
#pragma unroll
      for (int mi = 0; mi < 2; ++mi)
#pragma unroll
        for (int ni = 0; ni < 2; ++ni)
          acc[mi][ni] = __builtin_amdgcn_wmma_f32_16x16x32_bf16(
              false, a[mi].v, false, bf[ni].v, (short)0, acc[mi][ni],
              false, false);
    }
  }

  // ---- Epilogue: C layout (VGPR j -> M = j + 8*half, N = lane%16).
#pragma unroll
  for (int mi = 0; mi < 2; ++mi) {
#pragma unroll
    for (int ni = 0; ni < 2; ++ni) {
      const int l = l0 + n0w + ni * 16 + mrow;
#pragma unroll
      for (int j = 0; j < 8; ++j) {
        const int co = co0 + mi * 16 + j + 8 * half;
        out[((size_t)b * COUT + co) * LEN + l] = acc[mi][ni][j] + bias[co];
      }
    }
  }
}

extern "C" void kernel_launch(void* const* d_in, const int* in_sizes, int n_in,
                              void* d_out, int out_size, void* d_ws, size_t ws_size,
                              hipStream_t stream) {
  (void)in_sizes; (void)n_in; (void)out_size; (void)ws_size;
  const float* x      = (const float*)d_in[0];
  const float* w      = (const float*)d_in[1];
  const float* bias   = (const float*)d_in[2];
  const float* shifts = (const float*)d_in[3];

  unsigned short* w2 = (unsigned short*)d_ws;   // 9*128*384*2 = 884,736 B

  const int nw = KW * COUT * RTOT;
  prep_w_bf16<<<(nw + 255) / 256, 256, 0, stream>>>(w, w2);

  dim3 grid(LEN / NTILE, BATCH);                // 128 x 16 workgroups
  superonn_wmma<<<grid, 256, 0, stream>>>(x, w2, bias, shifts, (float*)d_out);
}
